// patch_head_48146583388360
// MI455X (gfx1250) — matrix operand order repaired and checked by reading, not compiled, not run on hardware
//
#include <hip/hip_runtime.h>

typedef _Float16 v4h  __attribute__((ext_vector_type(4)));
typedef _Float16 v8h  __attribute__((ext_vector_type(8)));
typedef _Float16 v16h __attribute__((ext_vector_type(16)));
typedef float    v4f  __attribute__((ext_vector_type(4)));
typedef float    v8f  __attribute__((ext_vector_type(8)));
typedef int      v4i  __attribute__((ext_vector_type(4)));

#define NPATCH 196
#define EMBED  768
#define KNBR   8
#define KTOP   4
#define ROWP   776          // padded LDS row stride in halfs (388 dwords -> conflict-free)
#define NTILE  13           // ceil(196/16)
#define BATCH  256

__global__ __launch_bounds__(1024)
void patch_head_gram_kernel(const float* __restrict__ x,
                            const int*   __restrict__ local_idx,
                            float* __restrict__ out_top,
                            float* __restrict__ out_loc)
{
    extern __shared__ char smem[];
    _Float16* xn    = (_Float16*)smem;                                  // [196][776] f16
    float*    sim   = (float*)(smem + NPATCH * ROWP * 2);               // [196][8]
    int*      lidx  = (int*)((char*)sim  + NPATCH * KNBR * 4);          // [196][8]
    int*      tsel  = (int*)((char*)lidx + NPATCH * KNBR * 4);          // [196][4]
    char*     tmask = (char*)tsel + NPATCH * KTOP * 4;                  // [169]

    const int b    = blockIdx.x;
    const int tid  = threadIdx.x;
    const int lane = tid & 31;
    const int wave = tid >> 5;
    const int nwav = blockDim.x >> 5;   // 32 waves

    // ---------- Phase 0a: zero tile mask ----------
    if (tid < NTILE * NTILE) tmask[tid] = 0;
    __syncthreads();

    // ---------- Phase 0b: local_idx -> LDS, mark needed Gram tiles ----------
    for (int i = tid; i < NPATCH * KNBR; i += blockDim.x) {
        const int N = local_idx[i];
        lidx[i] = N;
        const int M = i >> 3;                         // i / KNBR
        tmask[(M >> 4) * NTILE + (N >> 4)] = 1;       // racy same-value store: benign
    }

    // ---------- Phase 1: normalize x[b] rows -> LDS f16 ----------
    const float* xb = x + (size_t)b * NPATCH * EMBED;
    for (int row = wave; row < NPATCH; row += nwav) {
        const v4f* src = (const v4f*)(xb + (size_t)row * EMBED);
        v4f v[6];
        float ss = 0.f;
        #pragma unroll
        for (int i = 0; i < 6; ++i) {
            v[i] = src[lane + i * 32];
            ss += v[i].x * v[i].x + v[i].y * v[i].y + v[i].z * v[i].z + v[i].w * v[i].w;
        }
        #pragma unroll
        for (int o = 16; o >= 1; o >>= 1)
            ss += __shfl_xor(ss, o, 32);
        const float sc = 1.0f / fmaxf(sqrtf(ss), 1e-12f);
        _Float16* dst = xn + row * ROWP;
        #pragma unroll
        for (int i = 0; i < 6; ++i) {
            const int c0 = (lane + i * 32) * 4;
            v4h h;
            h.x = (_Float16)(v[i].x * sc);
            h.y = (_Float16)(v[i].y * sc);
            h.z = (_Float16)(v[i].z * sc);
            h.w = (_Float16)(v[i].w * sc);
            *(v4h*)(dst + c0) = h;      // ds_store_b64, 8B aligned
        }
    }
    __syncthreads();

    // ---------- Phase 2: needed Gram tiles via WMMA, scatter neighbor sims ----------
    for (int t = wave; t < NTILE * NTILE; t += nwav) {
        if (!tmask[t]) continue;                      // ~70 of 169 tiles contain neighbor pairs
        const int m0 = (t / NTILE) << 4;
        const int n0 = (t % NTILE) << 4;
        int mr = m0 + (lane & 15); if (mr > NPATCH - 1) mr = NPATCH - 1;
        int nr = n0 + (lane & 15); if (nr > NPATCH - 1) nr = NPATCH - 1;
        const int ka = (lane >> 4) << 3;    // A: K interleave 0 / 8
        const int kb = (lane >> 4) << 3;    // B: K interleave 0 / 8
        const _Float16* arow = xn + mr * ROWP + ka;
        const _Float16* brow = xn + nr * ROWP + kb;

        v8f c = {};
        #pragma unroll 4
        for (int k = 0; k < EMBED; k += 32) {
            v8h a0 = *(const v8h*)(arow + k);        // K kb..kb+7
            v8h a1 = *(const v8h*)(arow + k + 16);   // K kb+16..kb+23
            v8h b0 = *(const v8h*)(brow + k);        // K kb..kb+7
            v8h b1 = *(const v8h*)(brow + k + 16);   // K kb+16..kb+23
            v16h A, B;
            #pragma unroll
            for (int i = 0; i < 8; ++i) {
                A[i] = a0[i]; A[i + 8] = a1[i];
                B[i] = b0[i]; B[i + 8] = b1[i];
            }
            c = __builtin_amdgcn_wmma_f32_16x16x32_f16(
                    false, A, false, B, (short)0, c, false, false);
        }

        // scatter C: lane holds N = n0+lane%16, rows M = m0 + r (+8 upper half-wave).
        // local_idx rows are sorted ascending (np.nonzero), so the slot of a
        // matching N is the count of neighbors with smaller index.
        const int N = n0 + (lane & 15);
        if (N < NPATCH) {
            const int mhi = (lane >> 4) << 3;
            #pragma unroll
            for (int r = 0; r < 8; ++r) {
                const int M = m0 + mhi + r;
                if (M < NPATCH) {
                    const v4i nb0 = *(const v4i*)(lidx + M * KNBR);
                    const v4i nb1 = *(const v4i*)(lidx + M * KNBR + 4);
                    int slot = 0;
                    bool match = false;
                    #pragma unroll
                    for (int j = 0; j < 4; ++j) {
                        match |= (nb0[j] == N);  slot += (nb0[j] < N) ? 1 : 0;
                        match |= (nb1[j] == N);  slot += (nb1[j] < N) ? 1 : 0;
                    }
                    if (match) sim[M * KNBR + slot] = c[r];
                }
            }
        }
    }
    __syncthreads();

    // ---------- Phase 3: per-patch top-4 (stable, lowest index on ties) ----------
    if (tid < NPATCH) {
        float s[KNBR];
        #pragma unroll
        for (int j = 0; j < KNBR; ++j) s[j] = sim[tid * KNBR + j];
        unsigned used = 0;
        #pragma unroll
        for (int k = 0; k < KTOP; ++k) {
            float best = -3.0e38f; int bi = 0;
            #pragma unroll
            for (int j = 0; j < KNBR; ++j) {
                const bool ok = !((used >> j) & 1u) && (s[j] > best);
                if (ok) { best = s[j]; bi = j; }
            }
            used |= 1u << bi;
            tsel[tid * KTOP + k] = bi;
            out_top[((size_t)b * NPATCH + tid) * KTOP + k] = (float)bi;
        }
    }
    __syncthreads();

    // ---------- Phase 4: gather winners, stream out (nontemporal) ----------
    for (int rr = wave; rr < NPATCH * KTOP; rr += nwav) {
        const int n   = rr >> 2;
        const int k   = rr & 3;
        const int sel = tsel[n * KTOP + k];
        const int sp  = lidx[n * KNBR + sel];
        const v4f* src = (const v4f*)(xb + (size_t)sp * EMBED);          // L2-hot
        v4f* dst = (v4f*)(out_loc + (((size_t)b * NPATCH + n) * KTOP + k) * EMBED);
        #pragma unroll
        for (int i = 0; i < 6; ++i) {
            v4f v = src[lane + i * 32];
            __builtin_nontemporal_store(v, dst + lane + i * 32);
        }
    }
}

extern "C" void kernel_launch(void* const* d_in, const int* in_sizes, int n_in,
                              void* d_out, int out_size, void* d_ws, size_t ws_size,
                              hipStream_t stream) {
    (void)in_sizes; (void)n_in; (void)out_size; (void)d_ws; (void)ws_size;
    const float* x    = (const float*)d_in[0];
    const int*   lidx = (const int*)d_in[1];
    float* out_top = (float*)d_out;                                   // B*N*4 indices (as f32)
    float* out_loc = (float*)d_out + (size_t)BATCH * NPATCH * KTOP;   // B*N*4*768 gathered rows

    const size_t shmem = (size_t)NPATCH * ROWP * 2      // x_norm f16
                       + (size_t)NPATCH * KNBR * 4      // sim
                       + (size_t)NPATCH * KNBR * 4      // local_idx
                       + (size_t)NPATCH * KTOP * 4      // top selections
                       + 172;                           // tile mask
    hipLaunchKernelGGL(patch_head_gram_kernel, dim3(BATCH), dim3(1024), shmem, stream,
                       x, lidx, out_top, out_loc);
}
